// Seq2Seq_74079595921854
// MI455X (gfx1250) — matrix operand order repaired and checked by reading, not compiled, not run on hardware
//
#include <hip/hip_runtime.h>
#include <hip/hip_bf16.h>

typedef __attribute__((ext_vector_type(16))) __bf16 bf16x16;
typedef __attribute__((ext_vector_type(8)))  float  v8f;

#define B_SZ   64
#define S_LEN  64
#define T_LEN  64
#define EMB_D  256
#define EH_D   512
#define DH_D   512
#define OUTV   32000

union bfvec { bf16x16 v; int4 q[2]; };

// ---- WMMA fragment loaders (wave32 layouts per CDNA5 ISA 7.12.2) ----
// A: 16x32 (MxK) bf16. lane<16 holds row M=lane, K = {0..7, 16..23};
//    lane>=16 holds row M=lane-16, K = {8..15, 24..31}.
__device__ __forceinline__ bf16x16 load_a_frag(const __bf16* A, int lda,
                                               int m_base, int k, int lane) {
  const int row = m_base + (lane & 15);
  const int kh  = (lane >> 4) << 3;            // 0 or 8
  const __bf16* p = A + (size_t)row * lda + k;
  bfvec u;
  u.q[0] = *reinterpret_cast<const int4*>(p + kh);
  u.q[1] = *reinterpret_cast<const int4*>(p + 16 + kh);
  return u.v;
}
// B: 32x16 (KxN) bf16 from row-major W (N,K). lane<16: col N=lane, K = {0..7, 16..23};
//    lane>=16: col N=lane-16, K = {8..15, 24..31}.
__device__ __forceinline__ bf16x16 load_b_frag(const __bf16* W, int ldw,
                                               int n_base, int k, int lane) {
  const int n  = n_base + (lane & 15);
  const int kk = k + ((lane >> 4) << 3);       // +0 or +8
  const __bf16* p = W + (size_t)n * ldw + kk;
  bfvec u;
  u.q[0] = *reinterpret_cast<const int4*>(p);
  u.q[1] = *reinterpret_cast<const int4*>(p + 16);
  return u.v;
}

// ---- generic WMMA GEMM body: C(MxN,f32) = A(MxK,bf16) * W(NxK,bf16)^T + bias
// Each wave computes one 16x64 tile (4 accumulators), software-pipelined:
// fragments for k+32 are fetched while WMMAs for k execute, so the loadcnt
// wait at the next iteration hides a full WMMA block of latency.
// Requires M%16==0, N%64==0, K%32==0. act: 0=none, 1=tanh. Cbf: optional dup.
__device__ __forceinline__ void gemm_body(const __bf16* __restrict__ A, int lda,
                                          const __bf16* __restrict__ W, int ldw,
                                          const float* __restrict__ bias,
                                          float* __restrict__ C, int ldc,
                                          __bf16* __restrict__ Cbf,
                                          int M, int N, int K, int act) {
  const int lane = threadIdx.x & 31;
  const int wave = threadIdx.x >> 5;
  const int ngroups = N >> 6;
  const int wg = blockIdx.x * (blockDim.x >> 5) + wave;
  if (wg >= (M >> 4) * ngroups) return;
  const int m_base = (wg / ngroups) << 4;
  const int n_base = (wg % ngroups) << 6;

  v8f acc0 = {}, acc1 = {}, acc2 = {}, acc3 = {};
  // prologue: fetch k=0 fragments
  bf16x16 a  = load_a_frag(A, lda, m_base, 0, lane);
  bf16x16 b0 = load_b_frag(W, ldw, n_base +  0, 0, lane);
  bf16x16 b1 = load_b_frag(W, ldw, n_base + 16, 0, lane);
  bf16x16 b2 = load_b_frag(W, ldw, n_base + 32, 0, lane);
  bf16x16 b3 = load_b_frag(W, ldw, n_base + 48, 0, lane);

  for (int k = 0; k < K; k += 32) {
    // branchless next-k (last iteration re-fetches the final tile: same data)
    int kn = k + 32;
    kn = (kn < K) ? kn : (K - 32);
    // stream weights ahead through L2 (global_prefetch_b8; fault-silent)
    __builtin_prefetch(W + (size_t)(n_base + (lane & 15)) * ldw + kn + 32, 0, 0);
    // issue next-iteration loads before consuming current fragments
    bf16x16 an  = load_a_frag(A, lda, m_base, kn, lane);
    bf16x16 bn0 = load_b_frag(W, ldw, n_base +  0, kn, lane);
    bf16x16 bn1 = load_b_frag(W, ldw, n_base + 16, kn, lane);
    bf16x16 bn2 = load_b_frag(W, ldw, n_base + 32, kn, lane);
    bf16x16 bn3 = load_b_frag(W, ldw, n_base + 48, kn, lane);
    acc0 = __builtin_amdgcn_wmma_f32_16x16x32_bf16(false, a, false, b0, (short)0, acc0, false, false);
    acc1 = __builtin_amdgcn_wmma_f32_16x16x32_bf16(false, a, false, b1, (short)0, acc1, false, false);
    acc2 = __builtin_amdgcn_wmma_f32_16x16x32_bf16(false, a, false, b2, (short)0, acc2, false, false);
    acc3 = __builtin_amdgcn_wmma_f32_16x16x32_bf16(false, a, false, b3, (short)0, acc3, false, false);
    a = an; b0 = bn0; b1 = bn1; b2 = bn2; b3 = bn3;
  }
  // C/D layout: lane<16 -> N=n_base+lane, M=m_base+i; lane>=16 -> M=m_base+8+i.
  const int ncol = lane & 15;
  const int mrow = m_base + ((lane >> 4) << 3);
  v8f accs[4] = {acc0, acc1, acc2, acc3};
#pragma unroll
  for (int f = 0; f < 4; ++f) {
    const int n = n_base + f * 16 + ncol;
    const float bv = bias ? bias[n] : 0.f;
#pragma unroll
    for (int i = 0; i < 8; ++i) {
      float vv = accs[f][i] + bv;
      if (act == 1) vv = tanhf(vv);
      C[(size_t)(mrow + i) * ldc + n] = vv;
      if (Cbf) Cbf[(size_t)(mrow + i) * ldc + n] = (__bf16)vv;
    }
  }
}

__global__ void gemm_bf16(const __bf16* A, int lda, const __bf16* W, int ldw,
                          const float* bias, float* C, int ldc, __bf16* Cbf,
                          int M, int N, int K, int act) {
  gemm_body(A, lda, W, ldw, bias, C, ldc, Cbf, M, N, K, act);
}

// One launch covers gi/gh of both encoder directions (blockIdx.z selects GEMM).
__global__ void enc_step_gemm(int s, const __bf16* emb_bf, const __bf16* henc_bf,
                              const __bf16* Wih_f, const __bf16* Whh_f,
                              const __bf16* Wih_b, const __bf16* Whh_b,
                              const float* bih_f, const float* bhh_f,
                              const float* bih_b, const float* bhh_b,
                              float* gi_f, float* gh_f, float* gi_b, float* gh_b) {
  const __bf16* A; const __bf16* W; const float* bias; float* C; int lda, K;
  switch (blockIdx.z) {
    case 0:  A = emb_bf + (size_t)s        * B_SZ * EMB_D; lda = EMB_D;  K = EMB_D; W = Wih_f; bias = bih_f; C = gi_f; break;
    case 1:  A = henc_bf;                                  lda = 1024;   K = EH_D;  W = Whh_f; bias = bhh_f; C = gh_f; break;
    case 2:  A = emb_bf + (size_t)(63 - s) * B_SZ * EMB_D; lda = EMB_D;  K = EMB_D; W = Wih_b; bias = bih_b; C = gi_b; break;
    default: A = henc_bf + EH_D;                           lda = 1024;   K = EH_D;  W = Whh_b; bias = bhh_b; C = gh_b; break;
  }
  gemm_body(A, lda, W, K, bias, C, 3 * EH_D, nullptr, B_SZ, 3 * EH_D, K, 0);
}

__device__ __forceinline__ float sigmoidf_(float x) { return 1.f / (1.f + __expf(-x)); }

// Encoder GRU gating for both directions; writes recurrent h (f32 + bf16) and enc_out.
__global__ void enc_gate(int s, const float* gi_f, const float* gh_f,
                         const float* gi_b, const float* gh_b,
                         float* h32, __bf16* henc_bf,
                         float* enc_out, __bf16* enc_out_bf) {
  const int gid = blockIdx.x * blockDim.x + threadIdx.x;  // 2*64*512
  const int dir = gid >> 15;
  const int rem = gid & 32767;
  const int b = rem >> 9, j = rem & 511;
  const float* gi = dir ? gi_b : gi_f;
  const float* gh = dir ? gh_b : gh_f;
  const size_t base = (size_t)b * 1536;
  const float r = sigmoidf_(gi[base + j]        + gh[base + j]);
  const float z = sigmoidf_(gi[base + 512 + j]  + gh[base + 512 + j]);
  const float n = tanhf(gi[base + 1024 + j] + r * gh[base + 1024 + j]);
  const size_t hidx = (size_t)dir * 32768 + (size_t)b * 512 + j;
  const float h = (1.f - z) * n + z * h32[hidx];
  h32[hidx] = h;
  henc_bf[(size_t)b * 1024 + dir * 512 + j] = (__bf16)h;
  const int ss = dir ? (63 - s) : s;
  const size_t oidx = ((size_t)ss * B_SZ + b) * 1024 + dir * 512 + j;
  enc_out[oidx] = h;
  enc_out_bf[oidx] = (__bf16)h;
}

// Attention: scores = tanh(hW + enc_proj(+battn)) . v ; softmax over S; weighted.
__global__ void attn_kernel(int t, const int* src, const int* trg,
                            const float* hcat, const float* enc_proj,
                            const float* enc_out, const float* v0,
                            const float* dec_emb, float* weighted, __bf16* x_bf) {
  const int b = blockIdx.x;
  const int tid = threadIdx.x;
  __shared__ float part[256];
  __shared__ float sc[64];
  const int s = tid >> 2, seg = tid & 3;
  const float* hw = hcat + (size_t)b * 2048;
  const float* ep = enc_proj + ((size_t)s * B_SZ + b) * 512;
  float p = 0.f;
  for (int j = seg * 128; j < seg * 128 + 128; ++j)
    p += tanhf(hw[j] + ep[j]) * v0[j];
  part[tid] = p;
  __syncthreads();
  if (seg == 0) sc[s] = part[tid] + part[tid + 1] + part[tid + 2] + part[tid + 3];
  __syncthreads();
  if (tid < 32) {   // wave32 softmax over 64 scores
    float a0 = sc[tid], a1 = sc[tid + 32];
    float m = fmaxf(a0, a1);
    for (int o = 16; o; o >>= 1) m = fmaxf(m, __shfl_xor(m, o, 32));
    float e0 = __expf(a0 - m), e1 = __expf(a1 - m);
    float su = e0 + e1;
    for (int o = 16; o; o >>= 1) su += __shfl_xor(su, o, 32);
    sc[tid] = e0 / su; sc[tid + 32] = e1 / su;
  }
  __syncthreads();
  for (int h = tid; h < 1024; h += 256) {
    float acc = 0.f;
    for (int s2 = 0; s2 < 64; ++s2)
      acc += sc[s2] * enc_out[((size_t)s2 * B_SZ + b) * 1024 + h];
    weighted[(size_t)b * 1024 + h] = acc;
    x_bf[(size_t)b * 1280 + 256 + h] = (__bf16)acc;   // x = [e, weighted]
  }
  const int tok = (t == 0) ? src[b] : trg[t * B_SZ + b];
  x_bf[(size_t)b * 1280 + tid] = (__bf16)dec_emb[(size_t)tok * EMB_D + tid];
}

// Decoder GRU gate + build z = [h2, weighted, e] in bf16 for the output GEMM.
__global__ void dec_gru(int t, const float* gi, const float* hcat,
                        float* h32, __bf16* hbf,
                        const float* weighted, const float* dec_emb,
                        const int* src, const int* trg, __bf16* z_bf) {
  const int gid = blockIdx.x * blockDim.x + threadIdx.x;  // 64*1792
  const int b = gid / 1792, j = gid % 1792;
  if (j < 512) {
    const float* gh = hcat + (size_t)b * 2048 + 512;      // gh = hcat cols 512..2047
    const size_t base = (size_t)b * 1536;
    const float r = sigmoidf_(gi[base + j]       + gh[j]);
    const float z = sigmoidf_(gi[base + 512 + j] + gh[512 + j]);
    const float n = tanhf(gi[base + 1024 + j] + r * gh[1024 + j]);
    const float h = (1.f - z) * n + z * h32[(size_t)b * 512 + j];
    h32[(size_t)b * 512 + j] = h;
    hbf[(size_t)b * 512 + j] = (__bf16)h;
    z_bf[(size_t)b * 1792 + j] = (__bf16)h;
  } else if (j < 1536) {
    z_bf[(size_t)b * 1792 + j] = (__bf16)weighted[(size_t)b * 1024 + (j - 512)];
  } else {
    const int tok = (t == 0) ? src[b] : trg[t * B_SZ + b];
    z_bf[(size_t)b * 1792 + j] = (__bf16)dec_emb[(size_t)tok * EMB_D + (j - 1536)];
  }
}

// f32 -> bf16 (strided source) converter: dst[r*cols+c] = bf(src[r*ld + off + c])
__global__ void cvt_bf16(const float* src, __bf16* dst, int rows, int cols,
                         int srcLd, int srcOff) {
  const size_t gid = (size_t)blockIdx.x * blockDim.x + threadIdx.x;
  if (gid >= (size_t)rows * cols) return;
  const int r = (int)(gid / cols), c = (int)(gid % cols);
  dst[gid] = (__bf16)src[(size_t)r * srcLd + srcOff + c];
}

__global__ void embed_src(const int* src, const float* enc_emb, __bf16* emb_bf) {
  const size_t gid = (size_t)blockIdx.x * blockDim.x + threadIdx.x;  // 64*64*256
  const int sb = (int)(gid >> 8), k = (int)(gid & 255);
  emb_bf[gid] = (__bf16)enc_emb[(size_t)src[sb] * EMB_D + k];
}

__global__ void build_bias_cat(float* bias_cat, const float* bhh_d) {
  const int j = blockIdx.x * blockDim.x + threadIdx.x;  // 2048
  bias_cat[j] = (j < 512) ? 0.f : bhh_d[j - 512];
}

__global__ void fill_f32(float* p, size_t n) {
  const size_t gid = (size_t)blockIdx.x * blockDim.x + threadIdx.x;
  if (gid < n) p[gid] = 0.f;
}
__global__ void fill_bf16(__bf16* p, size_t n) {
  const size_t gid = (size_t)blockIdx.x * blockDim.x + threadIdx.x;
  if (gid < n) p[gid] = (__bf16)0.f;
}

extern "C" void kernel_launch(void* const* d_in, const int* in_sizes, int n_in,
                              void* d_out, int out_size, void* d_ws, size_t ws_size,
                              hipStream_t stream) {
  const int*   src     = (const int*)d_in[0];
  const int*   trg     = (const int*)d_in[1];
  const float* enc_emb = (const float*)d_in[2];
  const float* Wih_f   = (const float*)d_in[3];
  const float* Whh_f   = (const float*)d_in[4];
  const float* bih_f   = (const float*)d_in[5];
  const float* bhh_f   = (const float*)d_in[6];
  const float* Wih_b   = (const float*)d_in[7];
  const float* Whh_b   = (const float*)d_in[8];
  const float* bih_b   = (const float*)d_in[9];
  const float* bhh_b   = (const float*)d_in[10];
  const float* Wfc     = (const float*)d_in[11];
  const float* bfc     = (const float*)d_in[12];
  const float* dec_emb = (const float*)d_in[13];
  const float* Wattn   = (const float*)d_in[14];
  const float* battn   = (const float*)d_in[15];
  const float* v0      = (const float*)d_in[16];
  const float* Wih_d   = (const float*)d_in[17];
  const float* Whh_d   = (const float*)d_in[18];
  const float* bih_d   = (const float*)d_in[19];
  const float* bhh_d   = (const float*)d_in[20];
  const float* Wout    = (const float*)d_in[21];
  const float* bout    = (const float*)d_in[22];
  float* out = (float*)d_out;

  // ---- workspace carve-out (256B aligned) ----
  char* ws = (char*)d_ws;
  size_t off = 0;
  auto alloc = [&](size_t bytes) -> char* {
    off = (off + 255) & ~(size_t)255;
    char* p = ws + off; off += bytes; return p;
  };
  __bf16* Wihf_bf = (__bf16*)alloc((size_t)1536 * 256 * 2);
  __bf16* Whhf_bf = (__bf16*)alloc((size_t)1536 * 512 * 2);
  __bf16* Wihb_bf = (__bf16*)alloc((size_t)1536 * 256 * 2);
  __bf16* Whhb_bf = (__bf16*)alloc((size_t)1536 * 512 * 2);
  __bf16* Wfc_bf  = (__bf16*)alloc((size_t)512 * 1024 * 2);
  __bf16* Wae_bf  = (__bf16*)alloc((size_t)512 * 1024 * 2);
  __bf16* Wcat_bf = (__bf16*)alloc((size_t)2048 * 512 * 2);   // [Wa_h ; Whh_d]
  __bf16* Wihd_bf = (__bf16*)alloc((size_t)1536 * 1280 * 2);
  __bf16* Wout_bf = (__bf16*)alloc((size_t)OUTV * 1792 * 2);  // ~115 MB, L2-resident
  float*  bias_cat = (float*)alloc(2048 * 4);
  __bf16* emb_bf   = (__bf16*)alloc((size_t)S_LEN * B_SZ * EMB_D * 2);
  float*  h32enc   = (float*)alloc((size_t)2 * B_SZ * EH_D * 4);
  __bf16* henc_bf  = (__bf16*)alloc((size_t)B_SZ * 1024 * 2);
  float*  gi_f = (float*)alloc((size_t)B_SZ * 1536 * 4);
  float*  gh_f = (float*)alloc((size_t)B_SZ * 1536 * 4);
  float*  gi_b = (float*)alloc((size_t)B_SZ * 1536 * 4);
  float*  gh_b = (float*)alloc((size_t)B_SZ * 1536 * 4);
  float*  enc_out    = (float*)alloc((size_t)S_LEN * B_SZ * 1024 * 4);
  __bf16* enc_out_bf = (__bf16*)alloc((size_t)S_LEN * B_SZ * 1024 * 2);
  float*  enc_proj   = (float*)alloc((size_t)S_LEN * B_SZ * 512 * 4);
  float*  h_dec    = (float*)alloc((size_t)B_SZ * DH_D * 4);
  __bf16* h_dec_bf = (__bf16*)alloc((size_t)B_SZ * DH_D * 2);
  float*  hcat     = (float*)alloc((size_t)B_SZ * 2048 * 4);
  float*  weighted = (float*)alloc((size_t)B_SZ * 1024 * 4);
  __bf16* x_bf     = (__bf16*)alloc((size_t)B_SZ * 1280 * 2);
  float*  gi_d     = (float*)alloc((size_t)B_SZ * 1536 * 4);
  __bf16* z_bf     = (__bf16*)alloc((size_t)B_SZ * 1792 * 2);

  auto cvt = [&](const float* s_, __bf16* d_, int rows, int cols, int ld, int co) {
    size_t n = (size_t)rows * cols;
    cvt_bf16<<<dim3((unsigned)((n + 255) / 256)), dim3(256), 0, stream>>>(s_, d_, rows, cols, ld, co);
  };
  auto gemm = [&](const __bf16* A, int lda, const __bf16* W, int ldw,
                  const float* bias, float* C, int ldc, __bf16* Cbf,
                  int M, int N, int K, int act) {
    int waves = (M >> 4) * (N >> 6);
    gemm_bf16<<<dim3((unsigned)((waves + 3) >> 2)), dim3(128), 0, stream>>>(
        A, lda, W, ldw, bias, C, ldc, Cbf, M, N, K, act);
  };

  // ---- one-time conversion / init ----
  cvt(Wih_f, Wihf_bf, 1536, 256, 256, 0);
  cvt(Whh_f, Whhf_bf, 1536, 512, 512, 0);
  cvt(Wih_b, Wihb_bf, 1536, 256, 256, 0);
  cvt(Whh_b, Whhb_bf, 1536, 512, 512, 0);
  cvt(Wfc,   Wfc_bf,  512, 1024, 1024, 0);
  cvt(Wattn, Wae_bf,  512, 1024, 1536, 512);           // Wa_e = Wattn[:,512:]
  cvt(Wattn, Wcat_bf, 512,  512, 1536, 0);             // rows 0..511  = Wa_h
  cvt(Whh_d, Wcat_bf + (size_t)512 * 512, 1536, 512, 512, 0); // rows 512..2047
  cvt(Wih_d, Wihd_bf, 1536, 1280, 1280, 0);
  cvt(Wout,  Wout_bf, OUTV, 1792, 1792, 0);
  build_bias_cat<<<dim3(8), dim3(256), 0, stream>>>(bias_cat, bhh_d);
  embed_src<<<dim3((S_LEN * B_SZ * EMB_D) / 256), dim3(256), 0, stream>>>(src, enc_emb, emb_bf);
  fill_f32<<<dim3((2 * B_SZ * EH_D) / 256), dim3(256), 0, stream>>>(h32enc, (size_t)2 * B_SZ * EH_D);
  fill_bf16<<<dim3((B_SZ * 1024) / 256), dim3(256), 0, stream>>>(henc_bf, (size_t)B_SZ * 1024);
  fill_f32<<<dim3((B_SZ * OUTV) / 256), dim3(256), 0, stream>>>(out, (size_t)B_SZ * OUTV); // outputs[0]=0

  // ---- encoder: 64 sequential bidirectional GRU steps ----
  for (int s = 0; s < S_LEN; ++s) {
    enc_step_gemm<<<dim3(24, 1, 4), dim3(128), 0, stream>>>(
        s, emb_bf, henc_bf, Wihf_bf, Whhf_bf, Wihb_bf, Whhb_bf,
        bih_f, bhh_f, bih_b, bhh_b, gi_f, gh_f, gi_b, gh_b);
    enc_gate<<<dim3((2 * B_SZ * EH_D) / 256), dim3(256), 0, stream>>>(
        s, gi_f, gh_f, gi_b, gh_b, h32enc, henc_bf, enc_out, enc_out_bf);
  }

  // enc_proj[(s,b),:] = enc_out @ Wa_e^T + battn   (M=4096, N=512, K=1024)
  gemm(enc_out_bf, 1024, Wae_bf, 1024, battn, enc_proj, 512, nullptr,
       S_LEN * B_SZ, 512, 1024, 0);
  // hidden = tanh([hf,hb] @ Wfc^T + bfc)  -> h_dec (f32 + bf16)
  gemm(henc_bf, 1024, Wfc_bf, 1024, bfc, h_dec, 512, h_dec_bf, B_SZ, 512, 1024, 1);

  // ---- decoder: 63 sequential attention+GRU steps ----
  for (int t = 0; t < T_LEN - 1; ++t) {
    // hcat = h @ [Wa_h ; Whh_d]^T + [0 ; bhh_d]   (cols 0..511 = hW, 512..2047 = gh)
    gemm(h_dec_bf, 512, Wcat_bf, 512, bias_cat, hcat, 2048, nullptr, B_SZ, 2048, 512, 0);
    attn_kernel<<<dim3(B_SZ), dim3(256), 0, stream>>>(
        t, src, trg, hcat, enc_proj, enc_out, v0, dec_emb, weighted, x_bf);
    // gi = x @ Wih_d^T + bih_d   (K = 1280)
    gemm(x_bf, 1280, Wihd_bf, 1280, bih_d, gi_d, 1536, nullptr, B_SZ, 1536, 1280, 0);
    dec_gru<<<dim3((B_SZ * 1792) / 256), dim3(256), 0, stream>>>(
        t, gi_d, hcat, h_dec, h_dec_bf, weighted, dec_emb, src, trg, z_bf);
    // pred -> out[t+1]: z(64x1792) @ Wout^T(1792x32000) + bout
    gemm(z_bf, 1792, Wout_bf, 1792, bout, out + (size_t)(t + 1) * B_SZ * OUTV,
         OUTV, nullptr, B_SZ, OUTV, 1792, 0);
  }
}
